// AttnAutoEncoderRNN_55027120997036
// MI455X (gfx1250) — compile-verified, matrix operand order checked by reading, not run on hardware
//
#include <hip/hip_runtime.h>
#include <math.h>

#define DEV __device__ __forceinline__

typedef _Float16 v16h __attribute__((ext_vector_type(16)));
typedef _Float16 v8h  __attribute__((ext_vector_type(8)));
typedef float    v8f  __attribute__((ext_vector_type(8)));

constexpr int CV  = 32000;   // vocab
constexpr int CH  = 512;     // hidden
constexpr int CML = 160;     // max length
constexpr int CS  = 128;     // src len
constexpr int CT  = 128;     // tgt len
constexpr int CB  = 32;      // batch
constexpr int SOS = 1;
constexpr int H3  = 3 * CH;

// ---------------- WMMA fragment helpers (f16, 16x16x32) ----------------
// A fragment (16x32): lane L holds row m0+(L&15); elements 0..7 = K k0+8*half.. ,
// elements 8..15 = K k0+16+8*half.. (per CDNA5 ISA 16-bit A layout).
// B fragment mirrors A with lane = output column (weights stored row-major [N][K]).
DEV v16h load_frag_row(const _Float16* base, int ld, int r, int k0, int lane) {
  const int half8 = ((lane >> 4) & 1) * 8;
  const _Float16* p = base + (size_t)r * ld + k0 + half8;
  v8h lo = *(const v8h*)p;
  v8h hi = *(const v8h*)(p + 16);
  v16h out;
#pragma unroll
  for (int i = 0; i < 8; ++i) { out[i] = lo[i]; out[i + 8] = hi[i]; }
  return out;
}

DEV v8f wmma16(v16h a, v16h b, v8f c) {
  return __builtin_amdgcn_wmma_f32_16x16x32_f16(false, a, false, b, (short)0, c,
                                                false, false);
}

DEV float sigm(float x) { return 1.0f / (1.0f + __expf(-x)); }

// ---------------- f32 -> f16 convert ----------------
__global__ void k_cvt(const float* __restrict__ src, _Float16* __restrict__ dst, int n) {
  for (int i = blockIdx.x * blockDim.x + threadIdx.x; i < n; i += gridDim.x * blockDim.x)
    dst[i] = (_Float16)src[i];
}

// ---------------- embedding gather (+SOS/EOS handling), f16 output ------
// mode 0 (encoder): token(t,b) = t<CS ? toks[t*CB+b] : SOS   (steps = CS+1)
// mode 1 (decoder): token(t,b) = t==0 ? SOS : toks[(t-1)*CB+b] (steps = CT)
__global__ void k_gather(const int* __restrict__ toks, const float* __restrict__ emb,
                         _Float16* __restrict__ dst, int steps, int mode) {
  int total = steps * CB * CH;
  for (int idx = blockIdx.x * blockDim.x + threadIdx.x; idx < total;
       idx += gridDim.x * blockDim.x) {
    int h = idx % CH;
    int rb = idx / CH;
    int b = rb % CB;
    int t = rb / CB;
    int tok;
    if (mode == 0) tok = (t < CS) ? toks[t * CB + b] : SOS;
    else           tok = (t == 0) ? SOS : toks[(t - 1) * CB + b];
    dst[idx] = (_Float16)emb[(size_t)tok * CH + h];
  }
}

// ---------------- encoder input-side GEMM, all steps at once -----------
// gi[row][n] = encx[row][:] . Wih[n][:] + bih[n]; rows = (CS+1)*CB, n < 3H
__global__ void k_gi(const _Float16* __restrict__ encx, const _Float16* __restrict__ wih,
                     const float* __restrict__ bih, float* __restrict__ gi) {
  int lane = threadIdx.x & 31, wave = threadIdx.x >> 5;
  int m0 = blockIdx.x * 32;
  int nt = blockIdx.y * 8 + wave;   // gridDim.y = 12 -> 96 n-tiles
  int n0 = nt * 16;
  v8f acc0 = {}, acc1 = {};
  int rA = m0 + (lane & 15);
  int rB = n0 + (lane & 15);
  for (int k0 = 0; k0 < CH; k0 += 32) {
    v16h b  = load_frag_row(wih, CH, rB, k0, lane);
    v16h a0 = load_frag_row(encx, CH, rA, k0, lane);
    v16h a1 = load_frag_row(encx, CH, rA + 16, k0, lane);
    acc0 = wmma16(a0, b, acc0);
    acc1 = wmma16(a1, b, acc1);
  }
  float bias = bih[rB];
  int half8 = ((lane >> 4) & 1) * 8;
#pragma unroll
  for (int e = 0; e < 8; ++e) {
    int m = m0 + half8 + e;
    gi[(size_t)m * H3 + rB]        = acc0[e] + bias;
    gi[(size_t)(m + 16) * H3 + rB] = acc1[e] + bias;
  }
}

// ---------------- encoder recurrence (persistent, 1 WG) ----------------
// encT stored transposed [CH][CML] so it can be a WMMA B operand later.
__global__ void k_enc(const float* __restrict__ gi_all, const _Float16* __restrict__ whh,
                      const float* __restrict__ bhh, _Float16* __restrict__ encT,
                      _Float16* __restrict__ hout) {
  extern __shared__ char smem[];
  _Float16* hb0 = (_Float16*)smem;
  _Float16* hb1 = hb0 + CB * CH;
  int tid = threadIdx.x, lane = tid & 31, wave = tid >> 5;
  for (int i = tid; i < CB * CH * 2; i += 256) hb0[i] = (_Float16)0.f;
  for (int i = tid; i < CH * CML; i += 256) encT[i] = (_Float16)0.f;
  __syncthreads();
  int cur = 0;
  for (int t = 0; t <= CS; ++t) {
    const _Float16* hc = cur ? hb1 : hb0;
    _Float16* hn_ = cur ? hb0 : hb1;
    const float* gi = gi_all + (size_t)t * CB * H3;
    for (int task = wave; task < 64; task += 8) {
      int mt = task & 1, jt = task >> 1;
      int m0 = mt * 16;
      int rA = m0 + (lane & 15);
      int n = jt * 16 + (lane & 15);
      v8f ar = {}, az = {}, an = {};
      for (int k0 = 0; k0 < CH; k0 += 32) {
        v16h a = load_frag_row(hc, CH, rA, k0, lane);
        ar = wmma16(a, load_frag_row(whh, CH, n, k0, lane), ar);
        az = wmma16(a, load_frag_row(whh, CH, n + CH, k0, lane), az);
        an = wmma16(a, load_frag_row(whh, CH, n + 2 * CH, k0, lane), an);
      }
      float br = bhh[n], bz = bhh[n + CH], bn = bhh[n + 2 * CH];
      int half8 = ((lane >> 4) & 1) * 8;
#pragma unroll
      for (int e = 0; e < 8; ++e) {
        int m = m0 + half8 + e;
        const float* g = gi + (size_t)m * H3 + n;
        float r  = sigm(g[0] + ar[e] + br);
        float z  = sigm(g[CH] + az[e] + bz);
        float nv = tanhf(g[2 * CH] + r * (an[e] + bn));
        float hold = (float)hc[m * CH + n];
        float hv = (1.f - z) * nv + z * hold;
        hn_[m * CH + n] = (_Float16)hv;
        if (m == 0) encT[n * CML + t] = (_Float16)hv;  // batch-0 row only
      }
    }
    __syncthreads();
    cur ^= 1;
  }
  const _Float16* hf = cur ? hb1 : hb0;
  for (int i = tid; i < CB * CH; i += 256) hout[i] = hf[i];
}

// ---------------- decoder recurrence (persistent, 1 WG) ----------------
__global__ void k_dec(const _Float16* __restrict__ h0, const _Float16* __restrict__ decx,
                      const _Float16* __restrict__ attnW, const float* __restrict__ attnb,
                      const _Float16* __restrict__ encT, const _Float16* __restrict__ combW,
                      const float* __restrict__ combb, const _Float16* __restrict__ wih,
                      const _Float16* __restrict__ whh, const float* __restrict__ bih,
                      const float* __restrict__ bhh, _Float16* __restrict__ h2all) {
  extern __shared__ char smem[];
  char* p = smem;
  _Float16* hb0 = (_Float16*)p; p += CB * CH * 2;
  _Float16* hb1 = (_Float16*)p; p += CB * CH * 2;
  float*    sc  = (float*)p;    p += CB * CML * 4;
  _Float16* aw  = (_Float16*)p; p += CB * CML * 2;
  _Float16* app = (_Float16*)p; p += CB * CH * 2;
  _Float16* cmb = (_Float16*)p; p += CB * CH * 2;
  int tid = threadIdx.x, lane = tid & 31, wave = tid >> 5;
  int half8 = ((lane >> 4) & 1) * 8;
  for (int i = tid; i < CB * CH; i += 256) hb0[i] = h0[i];
  __syncthreads();
  int cur = 0;
  for (int t = 0; t < CT; ++t) {
    const _Float16* hc = cur ? hb1 : hb0;
    const _Float16* emb = decx + (size_t)t * CB * CH;
    // --- attention scores: [emb|h] @ attnW^T + b ---
    for (int task = wave; task < 20; task += 8) {
      int mt = task & 1, nt = task >> 1;
      int rA = mt * 16 + (lane & 15);
      int n = nt * 16 + (lane & 15);
      v8f acc = {};
      for (int k0 = 0; k0 < 2 * CH; k0 += 32) {
        v16h a = (k0 < CH) ? load_frag_row(emb, CH, rA, k0, lane)
                           : load_frag_row(hc, CH, rA, k0 - CH, lane);
        acc = wmma16(a, load_frag_row(attnW, 2 * CH, n, k0, lane), acc);
      }
      float bias = attnb[n];
#pragma unroll
      for (int e = 0; e < 8; ++e) sc[(mt * 16 + half8 + e) * CML + n] = acc[e] + bias;
    }
    __syncthreads();
    // --- row softmax -> aw (f16) ---
    if (tid < CB) {
      float mx = -1e30f;
      for (int n = 0; n < CML; ++n) mx = fmaxf(mx, sc[tid * CML + n]);
      float sum = 0.f;
      for (int n = 0; n < CML; ++n) sum += __expf(sc[tid * CML + n] - mx);
      float inv = 1.f / sum;
      for (int n = 0; n < CML; ++n)
        aw[tid * CML + n] = (_Float16)(__expf(sc[tid * CML + n] - mx) * inv);
    }
    __syncthreads();
    // --- applied = aw @ encoder_outputs (encT rows are output columns) ---
    for (int task = wave; task < 64; task += 8) {
      int mt = task & 1, nt = task >> 1;
      int rA = mt * 16 + (lane & 15);
      int n = nt * 16 + (lane & 15);
      v8f acc = {};
      for (int k0 = 0; k0 < CML; k0 += 32)
        acc = wmma16(load_frag_row(aw, CML, rA, k0, lane),
                     load_frag_row(encT, CML, n, k0, lane), acc);
#pragma unroll
      for (int e = 0; e < 8; ++e) app[(mt * 16 + half8 + e) * CH + n] = (_Float16)acc[e];
    }
    __syncthreads();
    // --- comb = relu([emb|applied] @ combW^T + b) ---
    for (int task = wave; task < 64; task += 8) {
      int mt = task & 1, nt = task >> 1;
      int rA = mt * 16 + (lane & 15);
      int n = nt * 16 + (lane & 15);
      v8f acc = {};
      for (int k0 = 0; k0 < 2 * CH; k0 += 32) {
        v16h a = (k0 < CH) ? load_frag_row(emb, CH, rA, k0, lane)
                           : load_frag_row(app, CH, rA, k0 - CH, lane);
        acc = wmma16(a, load_frag_row(combW, 2 * CH, n, k0, lane), acc);
      }
      float bias = combb[n];
#pragma unroll
      for (int e = 0; e < 8; ++e)
        cmb[(mt * 16 + half8 + e) * CH + n] = (_Float16)fmaxf(acc[e] + bias, 0.f);
    }
    __syncthreads();
    // --- GRU step ---
    _Float16* hn_ = cur ? hb0 : hb1;
    _Float16* h2 = h2all + (size_t)t * CB * CH;
    for (int task = wave; task < 64; task += 8) {
      int mt = task & 1, jt = task >> 1;
      int m0 = mt * 16;
      int rA = m0 + (lane & 15);
      int n = jt * 16 + (lane & 15);
      v8f ir = {}, iz = {}, in_ = {}, hr = {}, hz = {}, hn2 = {};
      for (int k0 = 0; k0 < CH; k0 += 32) {
        v16h ax = load_frag_row(cmb, CH, rA, k0, lane);
        v16h ah = load_frag_row(hc, CH, rA, k0, lane);
        ir  = wmma16(ax, load_frag_row(wih, CH, n, k0, lane), ir);
        iz  = wmma16(ax, load_frag_row(wih, CH, n + CH, k0, lane), iz);
        in_ = wmma16(ax, load_frag_row(wih, CH, n + 2 * CH, k0, lane), in_);
        hr  = wmma16(ah, load_frag_row(whh, CH, n, k0, lane), hr);
        hz  = wmma16(ah, load_frag_row(whh, CH, n + CH, k0, lane), hz);
        hn2 = wmma16(ah, load_frag_row(whh, CH, n + 2 * CH, k0, lane), hn2);
      }
      float bir = bih[n], biz = bih[n + CH], bin = bih[n + 2 * CH];
      float bhr = bhh[n], bhz = bhh[n + CH], bhn = bhh[n + 2 * CH];
#pragma unroll
      for (int e = 0; e < 8; ++e) {
        int m = m0 + half8 + e;
        float r  = sigm(ir[e] + bir + hr[e] + bhr);
        float z  = sigm(iz[e] + biz + hz[e] + bhz);
        float nv = tanhf(in_[e] + bin + r * (hn2[e] + bhn));
        float hold = (float)hc[m * CH + n];
        float hv = (1.f - z) * nv + z * hold;
        hn_[m * CH + n] = (_Float16)hv;
        h2[m * CH + n] = (_Float16)hv;
      }
    }
    __syncthreads();
    cur ^= 1;
  }
}

// ---------------- output projection + log_softmax (parallel over t) ----
__global__ void k_logits(const _Float16* __restrict__ h2all, const _Float16* __restrict__ outW,
                         const float* __restrict__ outb, float* __restrict__ out) {
  __shared__ __align__(16) _Float16 A[CB * CH];
  __shared__ float wmaxs[8][CB];
  __shared__ float wsums[8][CB];
  __shared__ float lse[CB];
  int t = blockIdx.x;
  int tid = threadIdx.x, lane = tid & 31, wave = tid >> 5;
  int half8 = ((lane >> 4) & 1) * 8;
  const _Float16* h2 = h2all + (size_t)t * CB * CH;
  for (int i = tid; i < CB * CH; i += 256) A[i] = h2[i];
  __syncthreads();
  float rm[16], rs[16];
#pragma unroll
  for (int i = 0; i < 16; ++i) { rm[i] = -1e30f; rs[i] = 0.f; }
  float* outt = out + (size_t)t * CB * CV;
  int rA0 = (lane & 15), rA1 = 16 + (lane & 15);
  for (int nt = wave; nt < CV / 16; nt += 8) {
    int n = nt * 16 + (lane & 15);
    v8f a0 = {}, a1 = {};
    for (int k0 = 0; k0 < CH; k0 += 32) {
      v16h b = load_frag_row(outW, CH, n, k0, lane);
      a0 = wmma16(load_frag_row(A, CH, rA0, k0, lane), b, a0);
      a1 = wmma16(load_frag_row(A, CH, rA1, k0, lane), b, a1);
    }
    float bias = outb[n];
#pragma unroll
    for (int e = 0; e < 8; ++e) {
      float v0 = a0[e] + bias, v1 = a1[e] + bias;
      outt[(size_t)(half8 + e) * CV + n] = v0;
      outt[(size_t)(16 + half8 + e) * CV + n] = v1;
      if (v0 > rm[e])     { rs[e]     = rs[e]     * __expf(rm[e] - v0) + 1.f;     rm[e] = v0; }
      else                  rs[e]    += __expf(v0 - rm[e]);
      if (v1 > rm[8 + e]) { rs[8 + e] = rs[8 + e] * __expf(rm[8 + e] - v1) + 1.f; rm[8 + e] = v1; }
      else                  rs[8 + e] += __expf(v1 - rm[8 + e]);
    }
  }
  // reduce (max,sumexp) across the 16 lanes that share rows
#pragma unroll
  for (int d = 1; d < 16; d <<= 1) {
#pragma unroll
    for (int s = 0; s < 16; ++s) {
      float om = __shfl_xor(rm[s], d, 32);
      float os = __shfl_xor(rs[s], d, 32);
      float mx = fmaxf(rm[s], om);
      rs[s] = rs[s] * __expf(rm[s] - mx) + os * __expf(om - mx);
      rm[s] = mx;
    }
  }
  if ((lane & 15) == 0) {
#pragma unroll
    for (int e = 0; e < 8; ++e) {
      wmaxs[wave][half8 + e] = rm[e];          wsums[wave][half8 + e] = rs[e];
      wmaxs[wave][16 + half8 + e] = rm[8 + e]; wsums[wave][16 + half8 + e] = rs[8 + e];
    }
  }
  __threadfence();
  __syncthreads();
  if (tid < CB) {
    float mx = -1e30f, sm = 0.f;
    for (int w = 0; w < 8; ++w) {
      float om = wmaxs[w][tid], os = wsums[w][tid];
      float m2 = fmaxf(mx, om);
      sm = sm * __expf(mx - m2) + os * __expf(om - m2);
      mx = m2;
    }
    lse[tid] = mx + __logf(sm);
  }
  __syncthreads();
  for (int m = 0; m < CB; ++m) {
    float l = lse[m];
    for (int n = tid; n < CV; n += 256)
      outt[(size_t)m * CV + n] -= l;
  }
}

// ======================= host launcher =======================
extern "C" void kernel_launch(void* const* d_in, const int* in_sizes, int n_in,
                              void* d_out, int out_size, void* d_ws, size_t ws_size,
                              hipStream_t stream) {
  (void)in_sizes; (void)n_in; (void)out_size; (void)ws_size;
  const int*   input_tensor  = (const int*)d_in[0];
  const int*   target_tensor = (const int*)d_in[1];
  const float* enc_emb = (const float*)d_in[2];
  const float* dec_emb = (const float*)d_in[3];
  const float* enc_Wih = (const float*)d_in[4];
  const float* enc_Whh = (const float*)d_in[5];
  const float* enc_bih = (const float*)d_in[6];
  const float* enc_bhh = (const float*)d_in[7];
  const float* dec_Wih = (const float*)d_in[8];
  const float* dec_Whh = (const float*)d_in[9];
  const float* dec_bih = (const float*)d_in[10];
  const float* dec_bhh = (const float*)d_in[11];
  const float* attn_W = (const float*)d_in[12];
  const float* attn_b = (const float*)d_in[13];
  const float* comb_W = (const float*)d_in[14];
  const float* comb_b = (const float*)d_in[15];
  const float* out_W  = (const float*)d_in[16];
  const float* out_b  = (const float*)d_in[17];
  float* out = (float*)d_out;

  // ---- workspace carve-out (all offsets 256B aligned) ----
  char* w = (char*)d_ws;
  auto carve = [&](size_t bytes) -> char* {
    char* r = w;
    w += (bytes + 255) & ~(size_t)255;
    return r;
  };
  _Float16* encWih16 = (_Float16*)carve((size_t)H3 * CH * 2);
  _Float16* encWhh16 = (_Float16*)carve((size_t)H3 * CH * 2);
  _Float16* decWih16 = (_Float16*)carve((size_t)H3 * CH * 2);
  _Float16* decWhh16 = (_Float16*)carve((size_t)H3 * CH * 2);
  _Float16* attnW16  = (_Float16*)carve((size_t)CML * 2 * CH * 2);
  _Float16* combW16  = (_Float16*)carve((size_t)CH * 2 * CH * 2);
  _Float16* outW16   = (_Float16*)carve((size_t)CV * CH * 2);
  _Float16* encx16   = (_Float16*)carve((size_t)(CS + 1) * CB * CH * 2);
  _Float16* decx16   = (_Float16*)carve((size_t)CT * CB * CH * 2);
  float*    gi_enc   = (float*)carve((size_t)(CS + 1) * CB * H3 * 4);
  _Float16* encT16   = (_Float16*)carve((size_t)CH * CML * 2);
  _Float16* henc16   = (_Float16*)carve((size_t)CB * CH * 2);
  _Float16* h2all16  = (_Float16*)carve((size_t)CT * CB * CH * 2);

  auto grid_for = [](int n) { int g = (n + 255) / 256; return g > 4096 ? 4096 : g; };

  // 1) weight conversion f32 -> f16
  k_cvt<<<grid_for(H3 * CH), 256, 0, stream>>>(enc_Wih, encWih16, H3 * CH);
  k_cvt<<<grid_for(H3 * CH), 256, 0, stream>>>(enc_Whh, encWhh16, H3 * CH);
  k_cvt<<<grid_for(H3 * CH), 256, 0, stream>>>(dec_Wih, decWih16, H3 * CH);
  k_cvt<<<grid_for(H3 * CH), 256, 0, stream>>>(dec_Whh, decWhh16, H3 * CH);
  k_cvt<<<grid_for(CML * 2 * CH), 256, 0, stream>>>(attn_W, attnW16, CML * 2 * CH);
  k_cvt<<<grid_for(CH * 2 * CH), 256, 0, stream>>>(comb_W, combW16, CH * 2 * CH);
  k_cvt<<<grid_for(CV * CH), 256, 0, stream>>>(out_W, outW16, CV * CH);

  // 2) embedding gathers
  k_gather<<<grid_for((CS + 1) * CB * CH), 256, 0, stream>>>(input_tensor, enc_emb,
                                                             encx16, CS + 1, 0);
  k_gather<<<grid_for(CT * CB * CH), 256, 0, stream>>>(target_tensor, dec_emb,
                                                       decx16, CT, 1);

  // 3) encoder input-side GEMM for all steps (parallel, WMMA)
  k_gi<<<dim3(CS + 1, 12), 256, 0, stream>>>(encx16, encWih16, enc_bih, gi_enc);

  // 4) encoder recurrence (persistent 1-WG, WMMA for h@Whh^T)
  k_enc<<<1, 256, 2 * CB * CH * 2, stream>>>(gi_enc, encWhh16, enc_bhh, encT16, henc16);

  // 5) decoder recurrence (persistent 1-WG; attention + context + comb + GRU)
  size_t dec_lds = (size_t)(2 * CB * CH * 2 + CB * CML * 4 + CB * CML * 2 +
                            CB * CH * 2 + CB * CH * 2);
  k_dec<<<1, 256, dec_lds, stream>>>(henc16, decx16, attnW16, attn_b, encT16,
                                     combW16, comb_b, decWih16, decWhh16,
                                     dec_bih, dec_bhh, h2all16);

  // 6) output projection + log_softmax, parallel over T (the heavy GEMM)
  k_logits<<<CT, 256, 0, stream>>>(h2all16, outW16, out_b, out);
}
